// MSA_34995393528035
// MI455X (gfx1250) — matrix operand order repaired and checked by reading, not compiled, not run on hardware
//
#include <hip/hip_runtime.h>
#include <hip/hip_bf16.h>

// ---------------------------------------------------------------------------
// AlphaFold3-style MSA/Pairformer block stack for MI455X (gfx1250, wave32).
// All GEMM-shaped math runs on V_WMMA_F32_16X16X32_BF16 (bf16 in, fp32 acc).
// fp32 tensors in workspace; bf16 conversion at fragment load.  The universal
// GEMM stages the shared A panel in LDS (one panel per tm-group per k-step)
// and prefetches the next B rows with global_prefetch.
// ---------------------------------------------------------------------------

#define S_DIM 192
#define N_RES 128
#define CM_   64
#define CZ_   128
#define CS_   256
#define HM_   8
#define HP_   4
#define DP_   32              // CZ_/HP_
#define TOKM  (S_DIM*N_RES)   // 24576
#define TOKZ  (S_DIM*S_DIM)   // 36864
#define OPM_TC 16             // OPM t-chunk

typedef __bf16 bf16_t;
typedef __attribute__((ext_vector_type(16))) __bf16 v16bf;
typedef __attribute__((ext_vector_type(8)))  float  v8f;

#define EPI_STORE        0   // D = acc (+bias)
#define EPI_ADD          1   // D += acc (+bias)
#define EPI_SIGMOID      2   // D = sigmoid(acc)
#define EPI_SILU         3   // D = silu(acc)
#define EPI_MUL          4   // D *= acc
#define EPI_SIGMOID_MUL  5   // D = sigmoid(acc) * D
#define EPI_GATE_ADD     6   // D += gate[o] * acc

__device__ __forceinline__ bf16_t to_bf(float x) { return (bf16_t)x; }

__device__ __forceinline__ v8f wmma32(const v16bf& a, const v16bf& b, const v8f& c) {
  return __builtin_amdgcn_wmma_f32_16x16x32_bf16(false, a, false, b, (short)0, c, false, false);
}

__device__ __forceinline__ float sigf(float x) { return 1.f / (1.f + __expf(-x)); }

// ---------------------------------------------------------------------------
// Universal WMMA GEMM: D[M,N] (op) A[M,K] @ W[K,N].  One wave per 16x16 tile,
// 8 waves per block, row-major tile order.  Waves sharing a tm row stage the
// 16xK A panel through LDS in 16x32 bf16 slabs.  Requires tiles % 8 == 0
// (true for every launch below) and N, K multiples of 16/32.
// rowmap: 0 direct; 1 transpose token (r=i*S+j -> j*S+i);
//         2 OPM chunk (r=s*TC+tl -> s*S + c0 + tl).
// ---------------------------------------------------------------------------
__global__ void __launch_bounds__(256)
gemm_bf16_wmma(const float* __restrict__ A, const float* __restrict__ W,
               const float* __restrict__ bias, const float* __restrict__ gate,
               float* __restrict__ D,
               int M, int N, int K, int epi, int rowmap, int rmC0)
{
  __shared__ bf16_t As[2][16][34];              // [group][row][k], padded
  const int lane = threadIdx.x & 31;
  const int wid  = threadIdx.x >> 5;
  const int tilesN = N >> 4;
  const int tiles  = (M >> 4) * tilesN;
  int id = blockIdx.x * 8 + wid;
  const bool active = id < tiles;
  if (id >= tiles) id = tiles - 1;              // clamp; barriers stay uniform
  const int tm = id / tilesN, tn = id % tilesN;
  const int G = (tilesN < 8) ? tilesN : 8;      // waves sharing one A panel
  const int group = (G == 8) ? 0 : (wid >> 2);  // G==4 -> two groups
  const int tgl = threadIdx.x - group * (G * 32);
  const int tmRow = tm * 16;
  const int bc  = tn * 16 + (lane & 15);
  const int kha = (lane >> 4) * 8;
  const int khb = (lane >> 4) * 16;
  const bf16_t* rowp = &As[group][lane & 15][0];

  v8f acc = {0.f,0.f,0.f,0.f,0.f,0.f,0.f,0.f};
  for (int k0 = 0; k0 < K; k0 += 32) {
    __syncthreads();
    for (int e = tgl; e < 512; e += G * 32) {   // cooperative A slab -> LDS
      const int r = e >> 5, kk = e & 31;
      As[group][r][kk] = to_bf(A[(size_t)(tmRow + r) * K + k0 + kk]);
      if (k0 + 32 < K)
        __builtin_prefetch(&A[(size_t)(tmRow + r) * K + k0 + 32 + kk], 0, 0);
    }
    __syncthreads();
    if (k0 + 32 < K)                            // prefetch next B rows
      __builtin_prefetch(&W[(size_t)(k0 + 32 + khb) * N + bc], 0, 0);
    v16bf af, bf;
#pragma unroll
    for (int v = 0; v < 8; ++v) {
      const int ka = khb + (v << 1);
      af[2*v]   = rowp[ka];
      af[2*v+1] = rowp[ka + 1];
      const int kb = k0 + khb + (v << 1);
      bf[2*v]   = to_bf(W[(size_t)kb * N + bc]);
      bf[2*v+1] = to_bf(W[(size_t)(kb + 1) * N + bc]);
    }
    acc = wmma32(af, bf, acc);
  }
  if (!active) return;
  const float bv = bias ? bias[bc] : 0.f;
#pragma unroll
  for (int r = 0; r < 8; ++r) {
    const int mm = r + ((lane >> 4) << 3);
    const int gr = tm * 16 + mm;
    int orow;
    if      (rowmap == 1) orow = (gr % S_DIM) * S_DIM + (gr / S_DIM);
    else if (rowmap == 2) orow = (gr >> 4) * S_DIM + rmC0 + (gr & 15);
    else                  orow = gr;
    const size_t o = (size_t)orow * N + bc;
    const float v = acc[r] + bv;
    switch (epi) {
      case EPI_STORE:       D[o] = v; break;
      case EPI_ADD:         D[o] += v; break;
      case EPI_SIGMOID:     D[o] = sigf(v); break;
      case EPI_SILU:        D[o] = v * sigf(v); break;
      case EPI_MUL:         D[o] *= v; break;
      case EPI_SIGMOID_MUL: D[o] = sigf(v) * D[o]; break;
      case EPI_GATE_ADD:    D[o] += gate[o] * v; break;
    }
  }
}

// ---------------------------------------------------------------------------
// Triangle multiplication einsum: per-channel batched GEMM.
// outgoing: t[i,j,c] = sum_k a[i,k,c] b[j,k,c]
// incoming: t[i,j,c] = sum_k a[k,i,c] b[k,j,c]
// ---------------------------------------------------------------------------
__global__ void __launch_bounds__(256)
trimul_wmma(const float* __restrict__ a, const float* __restrict__ b,
            float* __restrict__ t, int outgoing)
{
  const int lane = threadIdx.x & 31, wid = threadIdx.x >> 5;
  const int tS = S_DIM >> 4;                    // 12
  const int id = blockIdx.x * 8 + wid;
  if (id >= CZ_ * tS * tS) return;              // wave-uniform
  const int c = id / (tS * tS);
  const int rem = id % (tS * tS);
  const int ti = rem / tS, tj = rem % tS;
  const int ai = ti * 16 + (lane & 15);
  const int bj = tj * 16 + (lane & 15);
  const int kha = (lane >> 4) * 8, khb = (lane >> 4) * 16;
  v8f acc = {0.f,0.f,0.f,0.f,0.f,0.f,0.f,0.f};
  for (int k0 = 0; k0 < S_DIM; k0 += 32) {
    v16bf af, bf;
#pragma unroll
    for (int v = 0; v < 8; ++v) {
      const int ka = k0 + khb + (v << 1);
      if (outgoing) {
        af[2*v]   = to_bf(a[((size_t)ai * S_DIM + ka) * CZ_ + c]);
        af[2*v+1] = to_bf(a[((size_t)ai * S_DIM + ka + 1) * CZ_ + c]);
      } else {
        af[2*v]   = to_bf(a[((size_t)ka * S_DIM + ai) * CZ_ + c]);
        af[2*v+1] = to_bf(a[((size_t)(ka + 1) * S_DIM + ai) * CZ_ + c]);
      }
      const int kb = k0 + khb + (v << 1);
      if (outgoing) {
        bf[2*v]   = to_bf(b[((size_t)bj * S_DIM + kb) * CZ_ + c]);
        bf[2*v+1] = to_bf(b[((size_t)bj * S_DIM + kb + 1) * CZ_ + c]);
      } else {
        bf[2*v]   = to_bf(b[((size_t)kb * S_DIM + bj) * CZ_ + c]);
        bf[2*v+1] = to_bf(b[((size_t)(kb + 1) * S_DIM + bj) * CZ_ + c]);
      }
    }
    acc = wmma32(af, bf, acc);
  }
#pragma unroll
  for (int r = 0; r < 8; ++r) {
    const int mm = r + ((lane >> 4) << 3);
    const int gi = ti * 16 + mm;
    t[((size_t)gi * S_DIM + bj) * CZ_ + c] = acc[r];
  }
}

// ---------------------------------------------------------------------------
// OPM stage 1: op[s,tl,c,d] = (1/N) sum_n a[s,n,c] * b[t,n,d],  t = c0+tl.
// ab holds [token, 2*CM] with a = ch 0..CM-1, b = ch CM..2CM-1.
// ---------------------------------------------------------------------------
__global__ void __launch_bounds__(256)
opm1_wmma(const float* __restrict__ ab, float* __restrict__ op, int c0)
{
  const int lane = threadIdx.x & 31, wid = threadIdx.x >> 5;
  const int tpc = CM_ >> 4;                     // 4
  const int tpp = tpc * tpc;                    // 16
  const int id = blockIdx.x * 8 + wid;
  if (id >= S_DIM * OPM_TC * tpp) return;
  const int pair = id / tpp, rem = id % tpp;
  const int tc = rem / tpc, td = rem % tpc;
  const int s = pair / OPM_TC, tl = pair % OPM_TC;
  const int tg = c0 + tl;
  const int arow = tc * 16 + (lane & 15);
  const int bcol = td * 16 + (lane & 15);
  const int kha = (lane >> 4) * 8, khb = (lane >> 4) * 16;
  v8f acc = {0.f,0.f,0.f,0.f,0.f,0.f,0.f,0.f};
  for (int k0 = 0; k0 < N_RES; k0 += 32) {
    v16bf af, bf;
#pragma unroll
    for (int v = 0; v < 8; ++v) {
      const int ka = k0 + khb + (v << 1);
      af[2*v]   = to_bf(ab[((size_t)s * N_RES + ka) * (2*CM_) + arow]);
      af[2*v+1] = to_bf(ab[((size_t)s * N_RES + ka + 1) * (2*CM_) + arow]);
      const int kb = k0 + khb + (v << 1);
      bf[2*v]   = to_bf(ab[((size_t)tg * N_RES + kb) * (2*CM_) + CM_ + bcol]);
      bf[2*v+1] = to_bf(ab[((size_t)tg * N_RES + kb + 1) * (2*CM_) + CM_ + bcol]);
    }
    acc = wmma32(af, bf, acc);
  }
  const float sc = 1.f / (float)N_RES;
#pragma unroll
  for (int r = 0; r < 8; ++r) {
    const int mm = r + ((lane >> 4) << 3);
    const int gc = tc * 16 + mm;
    op[(((size_t)s * OPM_TC + tl) * CM_ + gc) * CM_ + bcol] = acc[r] * sc;
  }
}

// ---------------------------------------------------------------------------
// MSA pair-weighted averaging: o[i,n,h,d] = sum_j w[i,j,h] v[j,n,h,d].
// ---------------------------------------------------------------------------
__global__ void __launch_bounds__(256)
mpwa_attn_wmma(const float* __restrict__ w, const float* __restrict__ v,
               float* __restrict__ o)
{
  const int Dh = CM_ / HM_;                     // 8
  const int lane = threadIdx.x & 31, wid = threadIdx.x >> 5;
  const int tM = S_DIM >> 4;                    // 12
  const int tN = (N_RES * Dh) >> 4;             // 64
  const int id = blockIdx.x * 8 + wid;
  if (id >= HM_ * tM * tN) return;
  const int h = id / (tM * tN);
  const int rem = id % (tM * tN);
  const int tm = rem / tN, tn = rem % tN;
  const int ar = tm * 16 + (lane & 15);
  const int bcol = tn * 16 + (lane & 15);
  const int bn = bcol / Dh, bd = bcol % Dh;
  const int kha = (lane >> 4) * 8, khb = (lane >> 4) * 16;
  v8f acc = {0.f,0.f,0.f,0.f,0.f,0.f,0.f,0.f};
  for (int k0 = 0; k0 < S_DIM; k0 += 32) {
    v16bf af, bf;
#pragma unroll
    for (int vv = 0; vv < 8; ++vv) {
      const int ka = k0 + khb + (vv << 1);
      af[2*vv]   = to_bf(w[((size_t)ar * S_DIM + ka) * HM_ + h]);
      af[2*vv+1] = to_bf(w[((size_t)ar * S_DIM + ka + 1) * HM_ + h]);
      const int kb = k0 + khb + (vv << 1);
      bf[2*vv]   = to_bf(v[((size_t)kb * N_RES + bn) * CM_ + h * Dh + bd]);
      bf[2*vv+1] = to_bf(v[((size_t)(kb + 1) * N_RES + bn) * CM_ + h * Dh + bd]);
    }
    acc = wmma32(af, bf, acc);
  }
#pragma unroll
  for (int r = 0; r < 8; ++r) {
    const int mm = r + ((lane >> 4) << 3);
    const int gi = tm * 16 + mm;
    o[((size_t)gi * N_RES + bn) * CM_ + h * Dh + bd] = acc[r];
  }
}

// ---------------------------------------------------------------------------
// Triangle attention, one block per (head, i, j-tile).  QK^T via WMMA ->
// LDS, row softmax with pair bias, probs @ V via WMMA.
// ---------------------------------------------------------------------------
__global__ void __launch_bounds__(256)
triattn_wmma(const float* __restrict__ q, const float* __restrict__ k,
             const float* __restrict__ v, const float* __restrict__ bias,
             float* __restrict__ o, float scale)
{
  __shared__ float lg[16][208];
  const int lane = threadIdx.x & 31, wid = threadIdx.x >> 5;
  const int tJ = S_DIM >> 4;                    // 12
  const int h = blockIdx.x / (S_DIM * tJ);
  const int rem = blockIdx.x % (S_DIM * tJ);
  const int i = rem / tJ, jt = rem % tJ;
  const float* qh = q + (size_t)i * S_DIM * CZ_ + h * DP_;
  const float* kh = k + (size_t)i * S_DIM * CZ_ + h * DP_;
  const float* vh = v + (size_t)i * S_DIM * CZ_ + h * DP_;
  const int kha = (lane >> 4) * 8, khb = (lane >> 4) * 16;

  for (int kt = wid; kt < tJ; kt += 8) {        // QK^T, K = DP_ = 32
    const int jr = jt * 16 + (lane & 15);
    const int kc = kt * 16 + (lane & 15);
    v16bf af, bf;
#pragma unroll
    for (int vv = 0; vv < 8; ++vv) {
      const int ka = khb + (vv << 1);
      af[2*vv]   = to_bf(qh[(size_t)jr * CZ_ + ka]);
      af[2*vv+1] = to_bf(qh[(size_t)jr * CZ_ + ka + 1]);
      const int kb = khb + (vv << 1);
      bf[2*vv]   = to_bf(kh[(size_t)kc * CZ_ + kb]);
      bf[2*vv+1] = to_bf(kh[(size_t)kc * CZ_ + kb + 1]);
    }
    v8f acc = {0.f,0.f,0.f,0.f,0.f,0.f,0.f,0.f};
    acc = wmma32(af, bf, acc);
#pragma unroll
    for (int r = 0; r < 8; ++r) {
      const int mm = r + ((lane >> 4) << 3);
      const int jg = jt * 16 + mm;
      const int kg = kt * 16 + (lane & 15);
      lg[mm][kg] = acc[r] * scale + bias[((size_t)jg * S_DIM + kg) * HP_ + h];
    }
  }
  __syncthreads();
  if (threadIdx.x < 16) {                       // softmax over k
    const int rr = threadIdx.x;
    float mx = -1e30f;
    for (int kk = 0; kk < S_DIM; ++kk) mx = fmaxf(mx, lg[rr][kk]);
    float sm = 0.f;
    for (int kk = 0; kk < S_DIM; ++kk) { float e = __expf(lg[rr][kk] - mx); lg[rr][kk] = e; sm += e; }
    const float inv = 1.f / sm;
    for (int kk = 0; kk < S_DIM; ++kk) lg[rr][kk] *= inv;
  }
  __syncthreads();
  if (wid < (DP_ >> 4)) {                       // probs @ V, K = S
    const int arow = lane & 15;
    const int bcol = wid * 16 + (lane & 15);
    v8f acc = {0.f,0.f,0.f,0.f,0.f,0.f,0.f,0.f};
    for (int k0 = 0; k0 < S_DIM; k0 += 32) {
      v16bf af, bf;
#pragma unroll
      for (int vv = 0; vv < 8; ++vv) {
        const int ka = k0 + khb + (vv << 1);
        af[2*vv]   = to_bf(lg[arow][ka]);
        af[2*vv+1] = to_bf(lg[arow][ka + 1]);
        const int kb = k0 + khb + (vv << 1);
        bf[2*vv]   = to_bf(vh[(size_t)kb * CZ_ + bcol]);
        bf[2*vv+1] = to_bf(vh[(size_t)(kb + 1) * CZ_ + bcol]);
      }
      acc = wmma32(af, bf, acc);
    }
#pragma unroll
    for (int r = 0; r < 8; ++r) {
      const int mm = r + ((lane >> 4) << 3);
      o[((size_t)i * S_DIM + jt * 16 + mm) * CZ_ + h * DP_ + bcol] = acc[r];
    }
  }
}

// ----------------------------- small helpers -------------------------------

__global__ void __launch_bounds__(256)
ln_k(const float* __restrict__ x, const float* __restrict__ g,
     const float* __restrict__ b, float* __restrict__ y, int M, int C)
{
  const int wid = threadIdx.x >> 5, lane = threadIdx.x & 31;
  const int row = blockIdx.x * 8 + wid;
  if (row >= M) return;
  const float invC = 1.f / (float)C;
  const float* xr = x + (size_t)row * C;
  float s = 0.f;
  for (int c = lane; c < C; c += 32) s += xr[c];
#pragma unroll
  for (int o = 16; o > 0; o >>= 1) s += __shfl_xor(s, o, 32);
  const float mu = s * invC;
  float vs = 0.f;
  for (int c = lane; c < C; c += 32) { float d = xr[c] - mu; vs += d * d; }
#pragma unroll
  for (int o = 16; o > 0; o >>= 1) vs += __shfl_xor(vs, o, 32);
  const float inv = rsqrtf(vs * invC + 1e-5f);
  float* yr = y + (size_t)row * C;
  for (int c = lane; c < C; c += 32) yr[c] = (xr[c] - mu) * inv * g[c] + b[c];
}

__global__ void __launch_bounds__(256)
softmax_j_k(float* __restrict__ w)
{
  const int wid = threadIdx.x >> 5, lane = threadIdx.x & 31;
  const int row = blockIdx.x * 8 + wid;
  if (row >= S_DIM * HM_) return;
  const int i = row / HM_, h = row % HM_;
  float* base = w + (size_t)i * S_DIM * HM_ + h;
  float mx = -1e30f;
  for (int j = lane; j < S_DIM; j += 32) mx = fmaxf(mx, base[(size_t)j * HM_]);
#pragma unroll
  for (int o = 16; o > 0; o >>= 1) mx = fmaxf(mx, __shfl_xor(mx, o, 32));
  float sm = 0.f;
  for (int j = lane; j < S_DIM; j += 32) {
    float e = __expf(base[(size_t)j * HM_] - mx);
    base[(size_t)j * HM_] = e; sm += e;
  }
#pragma unroll
  for (int o = 16; o > 0; o >>= 1) sm += __shfl_xor(sm, o, 32);
  const float inv = 1.f / sm;
  for (int j = lane; j < S_DIM; j += 32) base[(size_t)j * HM_] *= inv;
}

__global__ void __launch_bounds__(256)
small_gemm_k(const float* __restrict__ A, const float* __restrict__ W,
             float* __restrict__ D, int M, int N, int K)
{
  const int idx = blockIdx.x * 256 + threadIdx.x;
  if (idx >= M * N) return;
  const int r = idx / N, c = idx % N;
  float s = 0.f;
  for (int k = 0; k < K; ++k) s += A[(size_t)r * K + k] * W[(size_t)k * N + c];
  D[idx] = s;
}

__global__ void __launch_bounds__(256)
m_init_k(const float* __restrict__ fmsa, const float* __restrict__ hd,
         const float* __restrict__ dv, const float* __restrict__ msaW,
         const float* __restrict__ ssing, float* __restrict__ m)
{
  const int idx = blockIdx.x * 256 + threadIdx.x;
  if (idx >= TOKM * CM_) return;
  const int tok = idx / CM_, c = idx % CM_;
  const int s = tok / N_RES;
  float acc = ssing[(size_t)s * CM_ + c];
  for (int qq = 0; qq < 23; ++qq) acc += fmsa[(size_t)tok * 23 + qq] * msaW[qq * CM_ + c];
  acc += hd[tok] * msaW[23 * CM_ + c] + dv[tok] * msaW[24 * CM_ + c];
  m[idx] = acc;
}

__global__ void __launch_bounds__(256)
eltmul_k(float* __restrict__ d, const float* __restrict__ a,
         const float* __restrict__ b, int n)
{
  const int idx = blockIdx.x * 256 + threadIdx.x;
  if (idx < n) d[idx] = a[idx] * b[idx];
}

__global__ void __launch_bounds__(256)
transpose_tok_k(const float* __restrict__ z, float* __restrict__ zt)
{
  const int idx = blockIdx.x * 256 + threadIdx.x;
  if (idx >= TOKZ * CZ_) return;
  const int c = idx % CZ_;
  const int t = idx / CZ_;
  const int i = t / S_DIM, j = t % S_DIM;
  zt[((size_t)j * S_DIM + i) * CZ_ + c] = z[idx];
}

// ---------------------------------------------------------------------------
// Host orchestration.
// Input order (setup_inputs insertion order, params depth-first insertion):
//  0 f_msa 1 f_has_del 2 f_del_val 3 s_inputs 4 z_ij
//  5 msa_W 6 single_W
//  7-11  opm: ln.g ln.b Wab Wo bo
//  12-19 mpwa: ln_m.g ln_m.b ln_z.g ln_z.b Wv Wb Wg Wo
//  20-24 trans_m: ln.g ln.b Wa Wb Wo
//  25-34 tri_out: ln_in.g ln_in.b Wa Wag Wb Wbg Wg ln_out.g ln_out.b Wo
//  35-44 tri_in (same layout)
//  45-52 attn_start: ln.g ln.b Wq Wk Wv Wb Wg Wo
//  53-60 attn_end (same layout)
//  61-65 trans_z: ln.g ln.b Wa Wb Wo
// ---------------------------------------------------------------------------
extern "C" void kernel_launch(void* const* d_in, const int* in_sizes, int n_in,
                              void* d_out, int out_size, void* d_ws, size_t ws_size,
                              hipStream_t stream)
{
  (void)in_sizes; (void)n_in; (void)out_size; (void)ws_size;
  auto P = [&](int i) { return reinterpret_cast<const float*>(d_in[i]); };
  float* ws = reinterpret_cast<float*>(d_ws);
  float* z  = reinterpret_cast<float*>(d_out);       // live pair state

  float* m    = ws;
  float* mt0  = m    + (size_t)TOKM * CM_;
  float* mt1  = mt0  + (size_t)TOKM * CM_;
  float* mt2  = mt1  + (size_t)TOKM * CM_;
  float* mh   = mt2  + (size_t)TOKM * CM_;           // TOKM*256
  float* zt0  = mh   + (size_t)TOKM * 256;
  float* zt1  = zt0  + (size_t)TOKZ * CZ_;
  float* zt2  = zt1  + (size_t)TOKZ * CZ_;
  float* zt3  = zt2  + (size_t)TOKZ * CZ_;
  float* zt4  = zt3  + (size_t)TOKZ * CZ_;
  float* zh   = zt4  + (size_t)TOKZ * CZ_;           // TOKZ*512
  float* wsm  = zh   + (size_t)TOKZ * 512;           // TOKZ*8
  float* ssing = wsm + (size_t)TOKZ * HM_;           // S*64

  auto gemm_full = [&](const float* A, const float* W, float* D,
                       int M, int N, int K, int epi,
                       const float* bias, const float* gate,
                       int rowmap, int rmC0) {
    const int tiles = (M / 16) * (N / 16);           // always % 8 == 0 here
    gemm_bf16_wmma<<<(tiles + 7) / 8, 256, 0, stream>>>(
        A, W, bias, gate, D, M, N, K, epi, rowmap, rmC0);
  };
  auto gemm = [&](const float* A, const float* W, float* D,
                  int M, int N, int K, int epi) {
    gemm_full(A, W, D, M, N, K, epi, nullptr, nullptr, 0, 0);
  };
  auto ln = [&](const float* x, float* y, const float* g, const float* b,
                int M, int C) {
    ln_k<<<(M + 7) / 8, 256, 0, stream>>>(x, g, b, y, M, C);
  };

  auto transition = [&](float* x, float* tmp, float* hid, int M, int C, int Ch, int pb) {
    ln(x, tmp, P(pb), P(pb + 1), M, C);
    gemm(tmp, P(pb + 2), hid, M, Ch, C, EPI_SILU);
    gemm(tmp, P(pb + 3), hid, M, Ch, C, EPI_MUL);
    gemm(hid, P(pb + 4), x, M, C, Ch, EPI_ADD);
  };

  auto trimult = [&](int pb, int outgoing) {
    ln(z, zt0, P(pb), P(pb + 1), TOKZ, CZ_);
    gemm(zt0, P(pb + 2), zt1, TOKZ, CZ_, CZ_, EPI_STORE);        // a value
    gemm(zt0, P(pb + 3), zt1, TOKZ, CZ_, CZ_, EPI_SIGMOID_MUL);  // a = sig(g)*a
    gemm(zt0, P(pb + 4), zt2, TOKZ, CZ_, CZ_, EPI_STORE);        // b value
    gemm(zt0, P(pb + 5), zt2, TOKZ, CZ_, CZ_, EPI_SIGMOID_MUL);  // b = sig(g)*b
    gemm(zt0, P(pb + 6), zt3, TOKZ, CZ_, CZ_, EPI_SIGMOID);      // out gate
    {
      const int tiles = CZ_ * (S_DIM / 16) * (S_DIM / 16);
      trimul_wmma<<<(tiles + 7) / 8, 256, 0, stream>>>(zt1, zt2, zt4, outgoing);
    }
    ln(zt4, zt0, P(pb + 7), P(pb + 8), TOKZ, CZ_);
    gemm_full(zt0, P(pb + 9), z, TOKZ, CZ_, CZ_, EPI_GATE_ADD, nullptr, zt3, 0, 0);
  };

  auto triattn = [&](int pb, int transposed) {
    const float* zsrc = z;
    if (transposed) {
      transpose_tok_k<<<(TOKZ * CZ_ + 255) / 256, 256, 0, stream>>>(z, zh);
      zsrc = zh;
    }
    ln(zsrc, zt0, P(pb), P(pb + 1), TOKZ, CZ_);
    gemm(zt0, P(pb + 2), zt1, TOKZ, CZ_, CZ_, EPI_STORE);        // q
    gemm(zt0, P(pb + 3), zt2, TOKZ, CZ_, CZ_, EPI_STORE);        // k
    gemm(zt0, P(pb + 4), zt3, TOKZ, CZ_, CZ_, EPI_STORE);        // v
    small_gemm_k<<<(TOKZ * HP_ + 255) / 256, 256, 0, stream>>>(
        zt0, P(pb + 5), wsm, TOKZ, HP_, CZ_);                    // pair bias
    gemm(zt0, P(pb + 6), zt4, TOKZ, CZ_, CZ_, EPI_SIGMOID);      // gate
    triattn_wmma<<<HP_ * S_DIM * (S_DIM / 16), 256, 0, stream>>>(
        zt1, zt2, zt3, wsm, zt0, 0.17677669529663687f);
    eltmul_k<<<(TOKZ * CZ_ + 255) / 256, 256, 0, stream>>>(zt0, zt0, zt4, TOKZ * CZ_);
    gemm_full(zt0, P(pb + 7), z, TOKZ, CZ_, CZ_, EPI_ADD,
              nullptr, nullptr, transposed ? 1 : 0, 0);
  };

  // ---- init: z = z_ij;  m = feat @ msa_W + (s_inputs @ single_W) broadcast
  hipMemcpyAsync(z, d_in[4], (size_t)TOKZ * CZ_ * sizeof(float),
                 hipMemcpyDeviceToDevice, stream);
  gemm(P(3), P(6), ssing, S_DIM, CM_, CS_, EPI_STORE);
  m_init_k<<<(TOKM * CM_ + 255) / 256, 256, 0, stream>>>(
      P(0), P(1), P(2), P(5), ssing, m);

  for (int blk = 0; blk < 2; ++blk) {
    // ---- outer product mean: z += reshape(a ⊗ b / N) @ Wo + bo
    ln(m, mt0, P(7), P(8), TOKM, CM_);
    gemm(mt0, P(9), mh, TOKM, 2 * CM_, CM_, EPI_STORE);          // ab
    for (int c0 = 0; c0 < S_DIM; c0 += OPM_TC) {
      const int tiles = S_DIM * OPM_TC * 16;
      opm1_wmma<<<(tiles + 7) / 8, 256, 0, stream>>>(mh, zh, c0);
      gemm_full(zh, P(10), z, S_DIM * OPM_TC, CZ_, CM_ * CM_, EPI_ADD,
                P(11), nullptr, 2, c0);
    }
    // ---- MSA pair-weighted averaging: m += (softmax_j(ln(z)Wb) · v ⊙ g) Wo
    ln(m, mt0, P(12), P(13), TOKM, CM_);
    ln(z, zt0, P(14), P(15), TOKZ, CZ_);
    small_gemm_k<<<(TOKZ * HM_ + 255) / 256, 256, 0, stream>>>(
        zt0, P(17), wsm, TOKZ, HM_, CZ_);
    softmax_j_k<<<(S_DIM * HM_ + 7) / 8, 256, 0, stream>>>(wsm);
    gemm(mt0, P(16), mt1, TOKM, CM_, CM_, EPI_STORE);            // v
    gemm(mt0, P(18), mt2, TOKM, CM_, CM_, EPI_SIGMOID);          // g
    {
      const int tiles = HM_ * (S_DIM / 16) * ((N_RES * (CM_ / HM_)) / 16);
      mpwa_attn_wmma<<<(tiles + 7) / 8, 256, 0, stream>>>(wsm, mt1, mt0);
    }
    eltmul_k<<<(TOKM * CM_ + 255) / 256, 256, 0, stream>>>(mt0, mt0, mt2, TOKM * CM_);
    gemm(mt0, P(19), m, TOKM, CM_, CM_, EPI_ADD);
    // ---- MSA transition
    transition(m, mt0, mh, TOKM, CM_, 4 * CM_, 20);
    // ---- triangle multiplications
    trimult(25, 1);
    trimult(35, 0);
    // ---- triangle attention (starting / ending node)
    triattn(45, 0);
    triattn(53, 1);
    // ---- pair transition
    transition(z, zt0, zh, TOKZ, CZ_, 4 * CZ_, 61);
  }
}
